// VSPN_6339371729327
// MI455X (gfx1250) — compile-verified, matrix operand order checked by reading, not run on hardware
//
#include <hip/hip_runtime.h>
#include <hip/hip_bf16.h>
#include <stdint.h>

// ---------------- problem constants ----------------
#define NN 50000
#define EE 600000
#define FF 64
#define HH 128
#define H3 384
#define GG 64
#define NH (NN * HH)   // 6,400,000

typedef __attribute__((ext_vector_type(16))) __bf16 v16bf;
typedef __attribute__((ext_vector_type(8)))  float  v8f;

// round-to-nearest-even f32 -> bf16 bits
__device__ __forceinline__ unsigned short f2bf(float f) {
  unsigned u = __float_as_uint(f);
  return (unsigned short)((u + 0x7FFFu + ((u >> 16) & 1u)) >> 16);
}

// fast tanh: hardware v_tanh_f32 on gfx1250 if the builtin exists
__device__ __forceinline__ float fast_tanh(float x) {
#if defined(__HIP_DEVICE_COMPILE__) && __has_builtin(__builtin_amdgcn_tanhf)
  return __builtin_amdgcn_tanhf(x);
#else
  float e = __expf(2.f * x);
  return 1.f - 2.f * __frcp_rn(e + 1.f);
#endif
}

__device__ __forceinline__ float fast_sigmoid(float x) {
  return __frcp_rn(1.f + __expf(-x));
}

// K index held by element e (0..15) of a 16-bit WMMA A/B operand register,
// for lane half khi = lane>>4  (ISA 7.12.2, 16-bit A-matrix 16x32 layout).
__device__ __forceinline__ int kmap(int khi, int e) {
  int i = e >> 1;
  int base = (i < 4) ? (khi * 8 + 2 * i) : (16 + khi * 8 + 2 * (i - 4));
  return base + (e & 1);
}

__device__ __forceinline__ v8f wmma_bf16(v16bf a, v16bf b, v8f c) {
#if defined(__HIP_DEVICE_COMPILE__)
  return __builtin_amdgcn_wmma_f32_16x16x32_bf16(false, a, false, b,
                                                 (short)0, c, false, false);
#else
  return c;
#endif
}

#if defined(__HIP_DEVICE_COMPILE__) &&                                     \
    __has_builtin(__builtin_amdgcn_global_load_async_to_lds_b128) &&       \
    __has_builtin(__builtin_amdgcn_s_wait_asynccnt)
#define USE_ASYNC_LDS 1
#endif

#ifdef USE_ASYNC_LDS
typedef __attribute__((ext_vector_type(4))) int v4i_t;
typedef __attribute__((address_space(1))) v4i_t glb_v4i_t;
typedef __attribute__((address_space(3))) v4i_t lds_v4i_t;
// one 16B async chunk: global -> LDS, then caller waits asynccnt
__device__ __forceinline__ void async_copy16(const void* gsrc, void* lds_dst) {
  glb_v4i_t* g = (glb_v4i_t*)(uintptr_t)gsrc;
  lds_v4i_t* l = (lds_v4i_t*)(unsigned)(uintptr_t)lds_dst;
  __builtin_amdgcn_global_load_async_to_lds_b128(g, l, 0, 0);
}
#endif

// ------------- pack f32 weight -> bf16 in WMMA B lane layout -------------
// Logical B is [K][Nout]; packed index = ((ct*nks + s)*32 + lane)*16 + e
// with k = s*32 + kmap(lane>>4, e), n = ct*16 + (lane&15).
// transposed==0: B[k][n] = src[k*Nout + n]   (W_l, W_in)
// transposed==1: B[k][n] = src[n*K + k]      (Wih^T, Whh^T from [Nout,K] src)
__global__ __launch_bounds__(256) void pack_weight_kernel(
    const float* __restrict__ src, unsigned short* __restrict__ dst,
    int K, int Nout, int transposed) {
  int nks = K >> 5;
  int total = (Nout >> 4) * nks * 512;
  int idx = blockIdx.x * 256 + threadIdx.x;
  if (idx >= total) return;
  int e = idx & 15;
  int l = (idx >> 4) & 31;
  int s = (idx >> 9) % nks;
  int ct = (idx >> 9) / nks;
  int k = s * 32 + kmap(l >> 4, e);
  int n = ct * 16 + (l & 15);
  float v = transposed ? src[n * K + k] : src[k * Nout + n];
  dst[idx] = f2bf(v);
}

// ------------- h = tanh(X @ W_in) ;  X [N,64] f32, W_in packed bf16 -------
__global__ __launch_bounds__(256) void embed_kernel(
    const float* __restrict__ X, const unsigned short* __restrict__ Bp,
    float* __restrict__ hf, unsigned short* __restrict__ hb) {
  __shared__ float rawA[16 * FF];                // 4 KB
  __shared__ unsigned short swzA[2 * 32 * 16];   // 2 KB
  int r0 = blockIdx.x * 16;
  int t = threadIdx.x;
  for (int i = t; i < 16 * FF; i += 256) rawA[i] = X[r0 * FF + i];
  __syncthreads();
  for (int idx = t; idx < 1024; idx += 256) {
    int e = idx & 15, l = (idx >> 4) & 31, s = idx >> 9;
    swzA[idx] = f2bf(rawA[(l & 15) * FF + s * 32 + kmap(l >> 4, e)]);
  }
  __syncthreads();
  int lane = t & 31, w = t >> 5;
  int ct = w;  // 8 column tiles of H=128, 8 waves
  v8f acc = {0.f, 0.f, 0.f, 0.f, 0.f, 0.f, 0.f, 0.f};
  for (int s = 0; s < 2; ++s) {
    v16bf a = *(const v16bf*)&swzA[(s * 32 + lane) * 16];
    v16bf b = *(const v16bf*)&Bp[((ct * 2 + s) * 32 + lane) * 16];
    acc = wmma_bf16(a, b, acc);
  }
  int khi = lane >> 4, col = ct * 16 + (lane & 15);
  for (int e = 0; e < 8; ++e) {
    int row = r0 + e + khi * 8;
    float v = fast_tanh(acc[e]);
    hf[row * HH + col] = v;
    hb[row * HH + col] = f2bf(v);
  }
}

// ------------- m = h @ W_l ;  h bf16 [N,128], W_l packed bf16 -------------
__global__ __launch_bounds__(256) void msg_kernel(
    const unsigned short* __restrict__ hb, const unsigned short* __restrict__ Bp,
    float* __restrict__ m) {
  __shared__ unsigned short rawA[16 * HH];       // 4 KB (bf16)
  __shared__ unsigned short swzA[4 * 32 * 16];   // 4 KB
  int r0 = blockIdx.x * 16;
  int t = threadIdx.x;
#ifdef USE_ASYNC_LDS
  async_copy16((const char*)(hb + (size_t)r0 * HH) + t * 16,
               (char*)rawA + t * 16);
  __builtin_amdgcn_s_wait_asynccnt(0);
#else
  ((uint4*)rawA)[t] = ((const uint4*)(hb + (size_t)r0 * HH))[t];
#endif
  __syncthreads();
  for (int idx = t; idx < 2048; idx += 256) {
    int e = idx & 15, l = (idx >> 4) & 31, s = idx >> 9;
    swzA[idx] = rawA[(l & 15) * HH + s * 32 + kmap(l >> 4, e)];
  }
  __syncthreads();
  int lane = t & 31, w = t >> 5;
  int ct = w;
  v8f acc = {0.f, 0.f, 0.f, 0.f, 0.f, 0.f, 0.f, 0.f};
  for (int s = 0; s < 4; ++s) {
    v16bf a = *(const v16bf*)&swzA[(s * 32 + lane) * 16];
    v16bf b = *(const v16bf*)&Bp[((ct * 4 + s) * 32 + lane) * 16];
    acc = wmma_bf16(a, b, acc);
  }
  int khi = lane >> 4, col = ct * 16 + (lane & 15);
  for (int e = 0; e < 8; ++e)
    m[(size_t)(r0 + e + khi * 8) * HH + col] = acc[e];
}

// ------------- fused gate GEMMs + GRU elementwise, h updated in place ----
__global__ __launch_bounds__(256) void gru_step_kernel(
    unsigned short* hb /*in+out bf16*/, float* hf /*in+out f32*/,
    const float* __restrict__ agg,
    const unsigned short* __restrict__ BihT, const unsigned short* __restrict__ BhhT,
    const float* __restrict__ bih, const float* __restrict__ bhh) {
  __shared__ char lds[57344];
  unsigned short* swzAgg = (unsigned short*)lds;            // 4 KB
  unsigned short* swzH   = (unsigned short*)(lds + 4096);   // 4 KB
  float* gi = (float*)(lds + 8192);                         // 24 KB [16][384]
  float* gh = (float*)(lds + 32768);                        // 24 KB [16][384]
  float* rawAgg = (float*)(lds + 8192);                     // 8 KB (dead before gi)
  unsigned short* rawH = (unsigned short*)(lds + 16384);    // 4 KB (dead before gi)
  int r0 = blockIdx.x * 16, t = threadIdx.x;
  {
    const float4* ga = (const float4*)(agg + (size_t)r0 * HH);
    float4* la = (float4*)rawAgg;
    la[t] = ga[t];
    la[t + 256] = ga[t + 256];
#ifdef USE_ASYNC_LDS
    async_copy16((const char*)(hb + (size_t)r0 * HH) + t * 16,
                 (char*)rawH + t * 16);
    __builtin_amdgcn_s_wait_asynccnt(0);
#else
    ((uint4*)rawH)[t] = ((const uint4*)(hb + (size_t)r0 * HH))[t];
#endif
  }
  __syncthreads();
  for (int idx = t; idx < 2048; idx += 256) {
    int e = idx & 15, l = (idx >> 4) & 31, s = idx >> 9;
    int k = s * 32 + kmap(l >> 4, e), row = l & 15;
    swzAgg[idx] = f2bf(rawAgg[row * HH + k]);
    swzH[idx]   = rawH[row * HH + k];
  }
  __syncthreads();
  int lane = t & 31, w = t >> 5, khi = lane >> 4, nloc = lane & 15;
  for (int j = w; j < 48; j += 8) {              // 24 gi tiles + 24 gh tiles
    int isGh = (j >= 24);
    int ct = isGh ? (j - 24) : j;
    const unsigned short* A = isGh ? swzH : swzAgg;
    const unsigned short* B = isGh ? BhhT : BihT;
    float* out = isGh ? gh : gi;
    v8f acc = {0.f, 0.f, 0.f, 0.f, 0.f, 0.f, 0.f, 0.f};
    for (int s = 0; s < 4; ++s) {
      v16bf av = *(const v16bf*)&A[(s * 32 + lane) * 16];
      v16bf bv = *(const v16bf*)&B[((ct * 4 + s) * 32 + lane) * 16];
      acc = wmma_bf16(av, bv, acc);
    }
    int col = ct * 16 + nloc;
    for (int e = 0; e < 8; ++e) out[(e + khi * 8) * H3 + col] = acc[e];
  }
  __syncthreads();
  for (int idx = t; idx < 2048; idx += 256) {
    int row = idx >> 7, c = idx & 127;
    float ir = gi[row * H3 + c]       + bih[c];
    float iz = gi[row * H3 + 128 + c] + bih[128 + c];
    float in_ = gi[row * H3 + 256 + c] + bih[256 + c];
    float hr = gh[row * H3 + c]       + bhh[c];
    float hz = gh[row * H3 + 128 + c] + bhh[128 + c];
    float hn = gh[row * H3 + 256 + c] + bhh[256 + c];
    float r = fast_sigmoid(ir + hr);
    float z = fast_sigmoid(iz + hz);
    float n = fast_tanh(in_ + r * hn);
    size_t p = (size_t)(r0 + row) * HH + c;
    float hnew = (1.f - z) * n + z * hf[p];
    hf[p] = hnew;
    hb[p] = f2bf(hnew);
  }
}

// ------------- agg[dst] += m[src] over edges ------------------------------
__global__ __launch_bounds__(256) void edge_scatter_kernel(
    const int* __restrict__ src, const int* __restrict__ dst,
    const float* __restrict__ m, float* __restrict__ agg, int nE) {
  int eIdx = blockIdx.x * 8 + (threadIdx.x >> 5);
  if (eIdx >= nE) return;
  int lane = threadIdx.x & 31;
  int s = src[eIdx], d = dst[eIdx];
  if (eIdx + 8 < nE)  // prefetch next edge's source row -> global_prefetch_b8
    __builtin_prefetch(m + (size_t)src[eIdx + 8] * HH + lane * 4, 0, 1);
  float4 v = ((const float4*)(m + (size_t)s * HH))[lane];
  float* out = agg + (size_t)d * HH + (size_t)lane * 4;
  atomicAdd(out + 0, v.x);
  atomicAdd(out + 1, v.y);
  atomicAdd(out + 2, v.z);
  atomicAdd(out + 3, v.w);
}

__global__ void zero_kernel(float* p, int n4) {
  int i = blockIdx.x * 256 + threadIdx.x;
  if (i < n4) ((float4*)p)[i] = make_float4(0.f, 0.f, 0.f, 0.f);
}

__global__ __launch_bounds__(256) void pool_accum_kernel(
    const float* __restrict__ hf, const int* __restrict__ batch,
    float* __restrict__ pool, float* __restrict__ cnt, int addCnt) {
  int nIdx = blockIdx.x * 8 + (threadIdx.x >> 5);
  if (nIdx >= NN) return;
  int lane = threadIdx.x & 31;
  int g = batch[nIdx];
  float4 v = ((const float4*)(hf + (size_t)nIdx * HH))[lane];
  float* out = pool + g * HH + lane * 4;
  atomicAdd(out + 0, v.x);
  atomicAdd(out + 1, v.y);
  atomicAdd(out + 2, v.z);
  atomicAdd(out + 3, v.w);
  if (addCnt && lane == 0) atomicAdd(&cnt[g], 1.0f);
}

__global__ __launch_bounds__(128) void head_kernel(
    const float* __restrict__ pa, const float* __restrict__ pv,
    const float* __restrict__ cnt, const float* __restrict__ predW,
    const float* __restrict__ predB, float* __restrict__ out) {
  __shared__ float red[128];
  int g = blockIdx.x, c = threadIdx.x;
  float cn = fmaxf(cnt[g], 1.0f);
  float a = fmaxf(pa[g * HH + c] / cn, 0.f);
  float v = fmaxf(pv[g * HH + c] / cn, 0.f);
  red[c] = a * predW[c] + v * predW[HH + c];
  __syncthreads();
  for (int s = 64; s > 0; s >>= 1) {
    if (c < s) red[c] += red[c + s];
    __syncthreads();
  }
  if (c == 0) out[g] = red[0] + predB[0];
}

// ---------------------------------------------------------------------------
extern "C" void kernel_launch(void* const* d_in, const int* in_sizes, int n_in,
                              void* d_out, int out_size, void* d_ws, size_t ws_size,
                              hipStream_t stream) {
  (void)in_sizes; (void)n_in; (void)out_size; (void)ws_size;
  const float* atom_x  = (const float*)d_in[0];
  const float* voro_x  = (const float*)d_in[1];
  const int*   atom_ei = (const int*)d_in[2];
  const int*   voro_ei = (const int*)d_in[3];
  const int*   batch   = (const int*)d_in[4];
  const float* W_atom_in = (const float*)d_in[5];
  const float* W_voro_in = (const float*)d_in[6];
  const float* atom_W  = (const float*)d_in[7];
  const float* voro_W  = (const float*)d_in[8];
  const float* atom_Wih = (const float*)d_in[9];
  const float* atom_Whh = (const float*)d_in[10];
  const float* atom_bih = (const float*)d_in[11];
  const float* atom_bhh = (const float*)d_in[12];
  const float* voro_Wih = (const float*)d_in[13];
  const float* voro_Whh = (const float*)d_in[14];
  const float* voro_bih = (const float*)d_in[15];
  const float* voro_bhh = (const float*)d_in[16];
  const float* pred_W  = (const float*)d_in[17];
  const float* pred_b  = (const float*)d_in[18];
  float* out = (float*)d_out;

  char* ws = (char*)d_ws;
  size_t off = 0;
  auto alloc = [&](size_t bytes) -> char* {
    off = (off + 255) & ~(size_t)255;
    char* p = ws + off;
    off += bytes;
    return p;
  };
  float* hf_a = (float*)alloc((size_t)NH * 4);
  float* hf_v = (float*)alloc((size_t)NH * 4);
  unsigned short* hb_a = (unsigned short*)alloc((size_t)NH * 2);
  unsigned short* hb_v = (unsigned short*)alloc((size_t)NH * 2);
  float* m_buf   = (float*)alloc((size_t)NH * 4);
  float* agg_buf = (float*)alloc((size_t)NH * 4);
  float* pool_a = (float*)alloc(GG * HH * 4);
  float* pool_v = (float*)alloc(GG * HH * 4);
  float* cnt    = (float*)alloc(GG * 4);
  unsigned short* Win_a  = (unsigned short*)alloc(8192 * 2);
  unsigned short* Win_v  = (unsigned short*)alloc(8192 * 2);
  unsigned short* Wl_a   = (unsigned short*)alloc(4 * 16384 * 2);
  unsigned short* Wl_v   = (unsigned short*)alloc(4 * 16384 * 2);
  unsigned short* WihT_a = (unsigned short*)alloc(49152 * 2);
  unsigned short* WhhT_a = (unsigned short*)alloc(49152 * 2);
  unsigned short* WihT_v = (unsigned short*)alloc(49152 * 2);
  unsigned short* WhhT_v = (unsigned short*)alloc(49152 * 2);

  // ---- pack all weights to bf16 WMMA layout ----
  pack_weight_kernel<<<(8192 + 255) / 256, 256, 0, stream>>>(W_atom_in, Win_a, FF, HH, 0);
  pack_weight_kernel<<<(8192 + 255) / 256, 256, 0, stream>>>(W_voro_in, Win_v, FF, HH, 0);
  for (int s = 0; s < 4; ++s) {
    pack_weight_kernel<<<64, 256, 0, stream>>>(atom_W + (size_t)s * HH * HH, Wl_a + (size_t)s * 16384, HH, HH, 0);
    pack_weight_kernel<<<64, 256, 0, stream>>>(voro_W + (size_t)s * HH * HH, Wl_v + (size_t)s * 16384, HH, HH, 0);
  }
  pack_weight_kernel<<<192, 256, 0, stream>>>(atom_Wih, WihT_a, HH, H3, 1);
  pack_weight_kernel<<<192, 256, 0, stream>>>(atom_Whh, WhhT_a, HH, H3, 1);
  pack_weight_kernel<<<192, 256, 0, stream>>>(voro_Wih, WihT_v, HH, H3, 1);
  pack_weight_kernel<<<192, 256, 0, stream>>>(voro_Whh, WhhT_v, HH, H3, 1);

  const int rowBlocks = NN / 16;           // 3125
  const int edgeBlocks = (EE + 7) / 8;     // 75000
  const int zeroNH = (NH / 4 + 255) / 256; // 6250

  struct Pass {
    const float* x; const unsigned short* Win;
    const unsigned short* Wl; const unsigned short* WihT; const unsigned short* WhhT;
    const float* bih; const float* bhh; const int* ei;
    float* hf; unsigned short* hb;
  } passes[2] = {
    {atom_x, Win_a, Wl_a, WihT_a, WhhT_a, atom_bih, atom_bhh, atom_ei, hf_a, hb_a},
    {voro_x, Win_v, Wl_v, WihT_v, WhhT_v, voro_bih, voro_bhh, voro_ei, hf_v, hb_v},
  };

  for (int p = 0; p < 2; ++p) {
    Pass& P = passes[p];
    embed_kernel<<<rowBlocks, 256, 0, stream>>>(P.x, P.Win, P.hf, P.hb);
    for (int s = 0; s < 4; ++s) {
      msg_kernel<<<rowBlocks, 256, 0, stream>>>(P.hb, P.Wl + (size_t)s * 16384, m_buf);
      zero_kernel<<<zeroNH, 256, 0, stream>>>(agg_buf, NH / 4);
      edge_scatter_kernel<<<edgeBlocks, 256, 0, stream>>>(P.ei, P.ei + EE, m_buf, agg_buf, EE);
      gru_step_kernel<<<rowBlocks, 256, 0, stream>>>(P.hb, P.hf, agg_buf, P.WihT, P.WhhT, P.bih, P.bhh);
    }
  }

  zero_kernel<<<8, 256, 0, stream>>>(pool_a, GG * HH / 4);
  zero_kernel<<<8, 256, 0, stream>>>(pool_v, GG * HH / 4);
  zero_kernel<<<1, 256, 0, stream>>>(cnt, GG / 4);
  pool_accum_kernel<<<(NN + 7) / 8, 256, 0, stream>>>(hf_a, batch, pool_a, cnt, 1);
  pool_accum_kernel<<<(NN + 7) / 8, 256, 0, stream>>>(hf_v, batch, pool_v, cnt, 0);
  head_kernel<<<GG, 128, 0, stream>>>(pool_a, pool_v, cnt, pred_W, pred_b, out);
}
